// TAG_LN_70574902608023
// MI455X (gfx1250) — matrix operand order repaired and checked by reading, not compiled, not run on hardware
//
#include <hip/hip_runtime.h>
#include <hip/hip_bf16.h>
#include <stdint.h>

#define NNODES 100000
#define NEDGES 1600000
#define NFEAT_IN 16
#define HID 64
#define OUTD 4

typedef __attribute__((ext_vector_type(16))) __bf16 v16bf;
typedef __attribute__((ext_vector_type(8)))  float  v8f;

union Frag16 {
  v16bf v;
  unsigned short s[16];
  uint4 q[2];
};

// ---------------- utility kernels ----------------

__global__ __launch_bounds__(256) void zero_f32(float* __restrict__ p, unsigned count4) {
  unsigned i = blockIdx.x * blockDim.x + threadIdx.x;
  if (i < count4) reinterpret_cast<float4*>(p)[i] = make_float4(0.f, 0.f, 0.f, 0.f);
}

__global__ __launch_bounds__(256) void degree_kernel(const long long* __restrict__ dst,
                                                     float* __restrict__ deg, unsigned nE) {
  unsigned e = blockIdx.x * blockDim.x + threadIdx.x;
  if (e < nE) atomicAdd(&deg[(int)dst[e]], 1.0f);
}

__global__ __launch_bounds__(256) void dinv_kernel(float* __restrict__ deg, unsigned n) {
  unsigned i = blockIdx.x * blockDim.x + threadIdx.x;
  if (i < n) {
    float d = deg[i];
    deg[i] = (d > 0.f) ? rsqrtf(d) : 0.f;
  }
}

// edge-parallel scatter-add: hout[dst] += dinv[src]*dinv[dst]*hin[src]
// one thread = one edge * one float4 chunk; F = 4<<SHIFT (compile-time)
template <int SHIFT>
__global__ __launch_bounds__(256) void prop_kernel(const long long* __restrict__ src,
                                                   const long long* __restrict__ dst,
                                                   const float* __restrict__ dinv,
                                                   const float* __restrict__ hin,
                                                   float* __restrict__ hout,
                                                   unsigned nE) {
  unsigned t = blockIdx.x * blockDim.x + threadIdx.x;
  unsigned e = t >> SHIFT;
  if (e >= nE) return;
  unsigned ch = t & ((1u << SHIFT) - 1u);
  constexpr int F = 4 << SHIFT;
  int s = (int)src[e];
  int d = (int)dst[e];
  float w = dinv[s] * dinv[d];
  float4 hv = *reinterpret_cast<const float4*>(hin + (size_t)s * F + ch * 4);
  float* o = hout + (size_t)d * F + ch * 4;
  atomicAdd(o + 0, w * hv.x);
  atomicAdd(o + 1, w * hv.y);
  atomicAdd(o + 2, w * hv.z);
  atomicAdd(o + 3, w * hv.w);
}

// W0 [4][16][64] f32 -> Bt0 [4][64 col][32 k] bf16 (k>=16 zero padded)
__global__ __launch_bounds__(256) void convert_w0(const float* __restrict__ W0,
                                                  unsigned short* __restrict__ Bt0) {
  int i = blockIdx.x * blockDim.x + threadIdx.x;
  if (i >= 4 * 64 * 32) return;
  int k = i & 31;
  int ch = i >> 5;
  int col = ch & 63;
  int hop = ch >> 6;
  float v = (k < NFEAT_IN) ? W0[((size_t)hop * NFEAT_IN + k) * HID + col] : 0.0f;
  __bf16 b = (__bf16)v;
  Bt0[i] = __builtin_bit_cast(unsigned short, b);
}

// Wh [3][4][64][64] f32 -> Bth [12][64 col][64 k] bf16
__global__ __launch_bounds__(256) void convert_wh(const float* __restrict__ Wh,
                                                  unsigned short* __restrict__ Bth) {
  int i = blockIdx.x * blockDim.x + threadIdx.x;
  if (i >= 12 * 64 * 64) return;
  int k = i & 63;
  int ch = i >> 6;
  int col = ch & 63;
  int lh = ch >> 6;  // layer*4 + hop
  __bf16 b = (__bf16)Wh[((size_t)lh * HID + k) * HID + col];
  Bth[i] = __builtin_bit_cast(unsigned short, b);
}

// -------- WMMA GEMM: C[n,64] (+)= A[n,FIN] * B  (bf16 x bf16 -> f32 acc) --------
// Bt layout: [col][KPAD] bf16 panels. One wave computes a 16x64 output strip:
// one A fragment per K-chunk feeds 4 column-tile WMMAs.
template <int FIN, int KPAD, int BETA, int RELU>
__global__ __launch_bounds__(256) void wmma_tag_gemm(const float* __restrict__ A,
                                                     const unsigned short* __restrict__ Bt,
                                                     const float* __restrict__ bias,
                                                     float* __restrict__ C,
                                                     int nRowTiles) {
  int wave = (int)((blockIdx.x * blockDim.x + threadIdx.x) >> 5);
  if (wave >= nRowTiles) return;              // wave-uniform; EXEC stays full
  int lane = threadIdx.x & 31;
  int group = lane >> 4;
  int r = lane & 15;
  int rowBase = wave * 16;

  v8f acc[4];
  if (BETA) {
#pragma unroll
    for (int ct = 0; ct < 4; ++ct)
#pragma unroll
      for (int j = 0; j < 8; ++j)
        acc[ct][j] = C[(size_t)(rowBase + j + 8 * group) * HID + ct * 16 + r];
  } else {
#pragma unroll
    for (int ct = 0; ct < 4; ++ct) {
      float bv = bias[ct * 16 + r];
#pragma unroll
      for (int j = 0; j < 8; ++j) acc[ct][j] = bv;
    }
  }

  const float* arow = A + (size_t)(rowBase + r) * FIN;

#pragma unroll
  for (int c0 = 0; c0 < KPAD; c0 += 32) {
    // A 16x32 bf16 fragment: lanes 0-15 hold K {c0+0..7, c0+16..23},
    // lanes 16-31 hold K {c0+8..15, c0+24..31} (row = lane&15).
    // Zero-fill test is lane-independent: 8*group never crosses a 16 boundary.
    Frag16 a;
#pragma unroll
    for (int h = 0; h < 2; ++h) {
      if (c0 + 16 * h < FIN) {
        int k0 = c0 + 16 * h + 8 * group;
        float4 f0 = *reinterpret_cast<const float4*>(arow + k0);
        float4 f1 = *reinterpret_cast<const float4*>(arow + k0 + 4);
        a.v[8 * h + 0] = (__bf16)f0.x;
        a.v[8 * h + 1] = (__bf16)f0.y;
        a.v[8 * h + 2] = (__bf16)f0.z;
        a.v[8 * h + 3] = (__bf16)f0.w;
        a.v[8 * h + 4] = (__bf16)f1.x;
        a.v[8 * h + 5] = (__bf16)f1.y;
        a.v[8 * h + 6] = (__bf16)f1.z;
        a.v[8 * h + 7] = (__bf16)f1.w;
      } else {
#pragma unroll
        for (int i = 0; i < 8; ++i) a.s[8 * h + i] = 0;
      }
    }
    // B 32x16 bf16 fragments: lane holds column ct*16+r; lanes 0-15 K {c0+0..7, c0+16..23},
    // lanes 16-31 K {c0+8..15, c0+24..31} -> two 8-bf16 runs from the [col][k] panel.
#pragma unroll
    for (int ct = 0; ct < 4; ++ct) {
      Frag16 b;
      const unsigned short* bp = Bt + (size_t)(ct * 16 + r) * KPAD + c0 + 8 * group;
      b.q[0] = *reinterpret_cast<const uint4*>(bp);
      b.q[1] = *reinterpret_cast<const uint4*>(bp + 16);
      acc[ct] = __builtin_amdgcn_wmma_f32_16x16x32_bf16(false, a.v, false, b.v,
                                                        (short)0, acc[ct], false, false);
    }
  }

#pragma unroll
  for (int ct = 0; ct < 4; ++ct)
#pragma unroll
    for (int j = 0; j < 8; ++j) {
      float v = acc[ct][j];
      if (RELU) v = fmaxf(v, 0.0f);
      C[(size_t)(rowBase + j + 8 * group) * HID + ct * 16 + r] = v;
    }
}

// ---------------- final FC: out[n,4] = h[n,64] @ fcW[64,4] + fcb ----------------
__global__ __launch_bounds__(256) void fc_kernel(const float* __restrict__ h,
                                                 const float* __restrict__ fcW,
                                                 const float* __restrict__ fcb,
                                                 float* __restrict__ out, unsigned n) {
  unsigned i = blockIdx.x * blockDim.x + threadIdx.x;
  if (i >= n) return;
  float acc0 = fcb[0], acc1 = fcb[1], acc2 = fcb[2], acc3 = fcb[3];
  const float* hr = h + (size_t)i * HID;
#pragma unroll
  for (int k = 0; k < HID; ++k) {
    float hv = hr[k];
    const float* wr = fcW + k * OUTD;
    acc0 += hv * wr[0];
    acc1 += hv * wr[1];
    acc2 += hv * wr[2];
    acc3 += hv * wr[3];
  }
  float* o = out + (size_t)i * OUTD;
  o[0] = acc0; o[1] = acc1; o[2] = acc2; o[3] = acc3;
}

// ---------------- host orchestration ----------------

extern "C" void kernel_launch(void* const* d_in, const int* in_sizes, int n_in,
                              void* d_out, int out_size, void* d_ws, size_t ws_size,
                              hipStream_t stream) {
  (void)in_sizes; (void)n_in; (void)out_size; (void)ws_size;
  const float*     x   = (const float*)d_in[0];
  const long long* ei  = (const long long*)d_in[1];   // int64 (2,E)
  const float*     W0  = (const float*)d_in[2];
  const float*     b0  = (const float*)d_in[3];
  const float*     Wh  = (const float*)d_in[4];
  const float*     bh  = (const float*)d_in[5];
  const float*     fcW = (const float*)d_in[6];
  const float*     fcb = (const float*)d_in[7];
  float*           out = (float*)d_out;
  const long long* src = ei;
  const long long* dst = ei + NEDGES;

  char* ws = (char*)d_ws;
  size_t off = 0;
  auto alloc = [&](size_t bytes) -> char* {
    char* p = ws + off;
    off = (off + bytes + 255) & ~(size_t)255;
    return p;
  };
  float* dinv = (float*)alloc((size_t)NNODES * 4);
  float* B1   = (float*)alloc((size_t)NNODES * HID * 4);
  float* B2   = (float*)alloc((size_t)NNODES * HID * 4);
  float* B3   = (float*)alloc((size_t)NNODES * HID * 4);
  unsigned short* Bt0 = (unsigned short*)alloc((size_t)4 * 64 * 32 * 2);
  unsigned short* Bth = (unsigned short*)alloc((size_t)12 * 64 * 64 * 2);

  dim3 blk(256);

  // GCN normalization
  zero_f32<<<(NNODES / 4 + 255) / 256, blk, 0, stream>>>(dinv, NNODES / 4);
  degree_kernel<<<(NEDGES + 255) / 256, blk, 0, stream>>>(dst, dinv, NEDGES);
  dinv_kernel<<<(NNODES + 255) / 256, blk, 0, stream>>>(dinv, NNODES);

  // bf16 transposed weight panels
  convert_w0<<<(4 * 64 * 32 + 255) / 256, blk, 0, stream>>>(W0, Bt0);
  convert_wh<<<(12 * 64 * 64 + 255) / 256, blk, 0, stream>>>(Wh, Bth);

  const int nRowTiles = NNODES / 16;                        // 6250 waves
  const int gemmBlocks = (nRowTiles * 32 + 255) / 256;

  auto gemm16 = [&](float* Cb, const float* A, const unsigned short* Bt,
                    const float* bias, int beta, int relu) {
    if (!beta)
      wmma_tag_gemm<16, 32, 0, 0><<<gemmBlocks, blk, 0, stream>>>(A, Bt, bias, Cb, nRowTiles);
    else if (relu)
      wmma_tag_gemm<16, 32, 1, 1><<<gemmBlocks, blk, 0, stream>>>(A, Bt, bias, Cb, nRowTiles);
    else
      wmma_tag_gemm<16, 32, 1, 0><<<gemmBlocks, blk, 0, stream>>>(A, Bt, bias, Cb, nRowTiles);
  };
  auto gemm64 = [&](float* Cb, const float* A, const unsigned short* Bt,
                    const float* bias, int beta, int relu) {
    if (!beta)
      wmma_tag_gemm<64, 64, 0, 0><<<gemmBlocks, blk, 0, stream>>>(A, Bt, bias, Cb, nRowTiles);
    else if (relu)
      wmma_tag_gemm<64, 64, 1, 1><<<gemmBlocks, blk, 0, stream>>>(A, Bt, bias, Cb, nRowTiles);
    else
      wmma_tag_gemm<64, 64, 1, 0><<<gemmBlocks, blk, 0, stream>>>(A, Bt, bias, Cb, nRowTiles);
  };
  auto prop16 = [&](float* ho, const float* hi) {
    unsigned c4 = (unsigned)NNODES * 16 / 4;
    zero_f32<<<(c4 + 255) / 256, blk, 0, stream>>>(ho, c4);
    unsigned nt = (unsigned)NEDGES << 2;
    prop_kernel<2><<<(nt + 255) / 256, blk, 0, stream>>>(src, dst, dinv, hi, ho, NEDGES);
  };
  auto prop64 = [&](float* ho, const float* hi) {
    unsigned c4 = (unsigned)NNODES * HID / 4;
    zero_f32<<<(c4 + 255) / 256, blk, 0, stream>>>(ho, c4);
    unsigned nt = (unsigned)NEDGES << 4;
    prop_kernel<4><<<(nt + 255) / 256, blk, 0, stream>>>(src, dst, dinv, hi, ho, NEDGES);
  };

  // ---- layer 0: fin = 16, out = relu(sum_k (A^k x) W0_k + b0) -> B1 ----
  gemm16(B1, x, Bt0 + 0 * 64 * 32, b0, 0, 0);
  prop16(B2, x);
  gemm16(B1, B2, Bt0 + 1 * 64 * 32, b0, 1, 0);
  prop16(B3, B2);
  gemm16(B1, B3, Bt0 + 2 * 64 * 32, b0, 1, 0);
  prop16(B2, B3);
  gemm16(B1, B2, Bt0 + 3 * 64 * 32, b0, 1, 1);  // + relu

  // ---- hidden layers 1..3 (Wh[0..2]); relu on first two only ----
  float* X = B1;  // current h
  float* Y = B2;  // next h / accumulator
  float* Z = B3;  // propagation scratch
  for (int l = 0; l < 3; ++l) {
    const unsigned short* Bt = Bth + (size_t)l * 4 * 64 * 64;
    const float* bias = bh + (size_t)l * HID;
    int relu = (l < 2) ? 1 : 0;
    gemm64(Y, X, Bt + 0 * 4096, bias, 0, 0);   // hop 0 (+bias)
    prop64(Z, X);                              // h1
    gemm64(Y, Z, Bt + 1 * 4096, bias, 1, 0);
    prop64(X, Z);                              // h2 (X no longer needed)
    gemm64(Y, X, Bt + 2 * 4096, bias, 1, 0);
    prop64(Z, X);                              // h3
    gemm64(Y, Z, Bt + 3 * 4096, bias, 1, relu);
    float* t = X; X = Y; Y = t;                // rotate: new h is in X
  }

  // ---- final FC ----
  fc_kernel<<<(NNODES + 255) / 256, blk, 0, stream>>>(X, fcW, fcb, out, NNODES);
}
